// RelationalGraphConvolution_42271068127502
// MI455X (gfx1250) — matrix operand order repaired and checked by reading, not compiled, not run on hardware
//
#include <hip/hip_runtime.h>
#include <hip/hip_bf16.h>
#include <stdint.h>

// ---------------------------------------------------------------------------
// Problem constants (match the reference)
// ---------------------------------------------------------------------------
#define NN      20000       // nodes
#define RR      11          // relations
#define BB      8           // bases
#define IN_DIM  300
#define KPAD    320         // 10 * 32 (WMMA f16 K-step = 32)
#define OUT_DIM 256
#define EE      640000      // edges

typedef __attribute__((ext_vector_type(16))) _Float16 v16h;
typedef __attribute__((ext_vector_type(8)))  float    v8f;

// ---------------------------------------------------------------------------
// Kernel 1: zero deg, init out with broadcast bias
// ---------------------------------------------------------------------------
__global__ void rgcn_init(float* __restrict__ deg, float* __restrict__ out,
                          const float* __restrict__ bias) {
    int idx = blockIdx.x * blockDim.x + threadIdx.x;     // 0 .. 5,119,999
    if (idx < RR * NN) deg[idx] = 0.0f;
    out[idx] = bias[idx & (OUT_DIM - 1)];
}

// ---------------------------------------------------------------------------
// Kernel 2: degree count per (rel, src) row
// ---------------------------------------------------------------------------
__global__ void rgcn_deg(const int* __restrict__ esrc, const int* __restrict__ erel,
                         float* __restrict__ deg) {
    int e = blockIdx.x * blockDim.x + threadIdx.x;
    atomicAdd(&deg[erel[e] * NN + esrc[e]], 1.0f);
}

// ---------------------------------------------------------------------------
// Kernel 3: convert features f32 -> f16, zero-pad K 300 -> 320
// ---------------------------------------------------------------------------
__global__ void rgcn_pack_feat(const float* __restrict__ feat,
                               _Float16* __restrict__ feat16) {
    int idx = blockIdx.x * blockDim.x + threadIdx.x;     // NN * KPAD threads
    int k = idx % KPAD;
    int n = idx / KPAD;
    feat16[idx] = (k < IN_DIM) ? (_Float16)feat[n * IN_DIM + k] : (_Float16)0.0f;
}

// ---------------------------------------------------------------------------
// Kernel 4: compute W[r] = sum_b comps[r,b]*bases[b] and pack B-fragments.
// Layout: wpack[r][nTile(16)][kStep(10)][lane(32)][elem(16)] so that each lane
// of the GEMM wave loads its 16 f16 B-operand values contiguously (32 bytes).
// B-element mapping per lane: half = lane>>4 selects K offset (+8),
// elem e -> K = kStep*32 + half*8 + (e&7) + (e>>3)*16 ;  col = nTile*16 + (lane&15).
// ---------------------------------------------------------------------------
__global__ void rgcn_pack_w(const float* __restrict__ comps,
                            const float* __restrict__ bases,
                            _Float16* __restrict__ wpack) {
    int idx = blockIdx.x * blockDim.x + threadIdx.x;     // 11*16*10*32*16 = 901,120
    int elem = idx & 15;
    int lane = (idx >> 4) & 31;
    int ks   = (idx >> 9) % 10;
    int rest = (idx >> 9) / 10;          // r*16 + nTile
    int nTile = rest & 15;
    int r     = rest >> 4;

    int half = lane >> 4;
    int k = ks * 32 + half * 8 + (elem & 7) + (elem >> 3) * 16;  // K index (0..319)
    int o = nTile * 16 + (lane & 15);                // output column

    float acc = 0.0f;
    if (k < IN_DIM) {
        #pragma unroll
        for (int b = 0; b < BB; ++b)
            acc += comps[r * BB + b] * bases[((size_t)b * IN_DIM + k) * OUT_DIM + o];
    }
    wpack[idx] = (_Float16)acc;
}

// ---------------------------------------------------------------------------
// Kernel 5: WMMA GEMM  T[r] = feat16 @ W[r]   (f16 in, f32 acc, f16 out)
// Each wave computes a 16(M) x 64(N) strip = 4 adjacent 16x16 tiles; the A
// fragment is loaded once per K-step and reused by 4 WMMAs. Operand fragments
// are explicitly double-buffered: K-step ks+1 loads are issued before the 4
// WMMAs of step ks, so the wave never drains loadcnt to 0 before a wmma.
// 11 * 1250 * 4 = 55,000 waves exactly (EXEC always full, as WMMA requires).
// ---------------------------------------------------------------------------
union Frag { v16h v; uint4 q[2]; };

__global__ __launch_bounds__(256) void rgcn_gemm(
    const _Float16* __restrict__ feat16,   // [NN][KPAD]
    const _Float16* __restrict__ wpack,    // packed fragments
    _Float16* __restrict__ T)              // [RR][NN][OUT_DIM]
{
    const int lane = threadIdx.x & 31;
    const int wave = threadIdx.x >> 5;
    int flat = blockIdx.x * 8 + wave;              // 0 .. 54,999 (exact grid)
    int r    = flat / (1250 * 4);
    int rem  = flat - r * (1250 * 4);
    int mTile  = rem >> 2;                         // 0..1249
    int nGroup = rem & 3;                          // 0..3 -> nTiles 4g..4g+3

    const int half = lane >> 4;
    const int m16  = lane & 15;

    // A: row-major f16 rows; per-lane fragment = two 16B chunks per K-step
    const _Float16* aRow = feat16 + (size_t)(mTile * 16 + m16) * KPAD + half * 8;
    // B: pre-packed fragments, 16 f16 per lane contiguous; per-nTile stride
    // = 10 kSteps * 32 lanes * 16 elems = 5120 halves
    const _Float16* bBase = wpack +
        (((size_t)(r * 16 + nGroup * 4) * 10) * 32 + lane) * 16;

    Frag a[2], b0[2], b1[2], b2[2], b3[2];

    auto loadStep = [&](int ks, int slot) {
        a[slot].q[0] = *(const uint4*)(aRow + ks * 32);        // K = half*8 + 0..7
        a[slot].q[1] = *(const uint4*)(aRow + ks * 32 + 16);   // K = +16
        const _Float16* bp = bBase + (size_t)ks * 512;
        b0[slot].q[0] = *(const uint4*)(bp);
        b0[slot].q[1] = *(const uint4*)(bp + 8);
        b1[slot].q[0] = *(const uint4*)(bp + 5120);
        b1[slot].q[1] = *(const uint4*)(bp + 5128);
        b2[slot].q[0] = *(const uint4*)(bp + 10240);
        b2[slot].q[1] = *(const uint4*)(bp + 10248);
        b3[slot].q[0] = *(const uint4*)(bp + 15360);
        b3[slot].q[1] = *(const uint4*)(bp + 15368);
    };

    loadStep(0, 0);

    v8f c0 = {}, c1 = {}, c2 = {}, c3 = {};
    #pragma unroll
    for (int ks = 0; ks < 10; ++ks) {
        const int cur = ks & 1;
        const int nxt = cur ^ 1;
        if (ks < 9) loadStep(ks + 1, nxt);     // prefetch next K-step
        c0 = __builtin_amdgcn_wmma_f32_16x16x32_f16(false, a[cur].v, false,
                 b0[cur].v, (short)0, c0, false, false);
        c1 = __builtin_amdgcn_wmma_f32_16x16x32_f16(false, a[cur].v, false,
                 b1[cur].v, (short)0, c1, false, false);
        c2 = __builtin_amdgcn_wmma_f32_16x16x32_f16(false, a[cur].v, false,
                 b2[cur].v, (short)0, c2, false, false);
        c3 = __builtin_amdgcn_wmma_f32_16x16x32_f16(false, a[cur].v, false,
                 b3[cur].v, (short)0, c3, false, false);
    }

    // D layout: VGPR v -> row (v + half*8), lane&15 -> column
    _Float16* outp = T + (((size_t)r * NN) + mTile * 16 + half * 8) * OUT_DIM
                       + nGroup * 64 + m16;
    #pragma unroll
    for (int v = 0; v < 8; ++v) {
        _Float16* row = outp + (size_t)v * OUT_DIM;
        row[0]  = (_Float16)c0[v];
        row[16] = (_Float16)c1[v];
        row[32] = (_Float16)c2[v];
        row[48] = (_Float16)c3[v];
    }
}

// ---------------------------------------------------------------------------
// Kernel 6: edge scatter  out[src] += (1/deg[rel*N+src]) * T[rel][dst]
// One wave per edge; lane l covers 8 contiguous outputs -> fully coalesced
// 512B row gather + 8 global_atomic_add_f32 per lane into L2-resident out.
// ---------------------------------------------------------------------------
__global__ __launch_bounds__(256) void rgcn_scatter(
    const int* __restrict__ esrc, const int* __restrict__ erel,
    const int* __restrict__ edst, const float* __restrict__ deg,
    const _Float16* __restrict__ T, float* __restrict__ out)
{
    const int lane = threadIdx.x & 31;
    int e = blockIdx.x * 8 + (threadIdx.x >> 5);   // 80,000 blocks * 8 = EE exact
    int s = esrc[e];
    int r = erel[e];
    int d = edst[e];
    float val = 1.0f / deg[r * NN + s];

    union { uint4 q; _Float16 h[8]; } t;
    t.q = *(const uint4*)(T + ((size_t)r * NN + d) * OUT_DIM + lane * 8);

    float* orow = out + (size_t)s * OUT_DIM + lane * 8;
    #pragma unroll
    for (int j = 0; j < 8; ++j)
        atomicAdd(&orow[j], val * (float)t.h[j]);
}

// ---------------------------------------------------------------------------
// Launch
// ---------------------------------------------------------------------------
extern "C" void kernel_launch(void* const* d_in, const int* in_sizes, int n_in,
                              void* d_out, int out_size, void* d_ws, size_t ws_size,
                              hipStream_t stream) {
    const float* features = (const float*)d_in[0];   // [NN*IN_DIM]
    const float* comps    = (const float*)d_in[1];   // [RR*BB]
    const float* bases    = (const float*)d_in[2];   // [BB*IN_DIM*OUT_DIM]
    const float* bias     = (const float*)d_in[3];   // [OUT_DIM]
    const int*   esrc     = (const int*)d_in[4];
    const int*   erel     = (const int*)d_in[5];
    const int*   edst     = (const int*)d_in[6];
    float* out = (float*)d_out;                      // [NN*OUT_DIM] f32

    // Workspace layout (256B aligned offsets)
    char* ws = (char*)d_ws;
    _Float16* feat16 = (_Float16*)(ws);                         // 12,800,000 B
    _Float16* wpack  = (_Float16*)(ws + 12800000);              //  1,802,240 B
    float*    deg    = (float*)   (ws + 14602240);              //    880,000 B
    _Float16* T      = (_Float16*)(ws + 15482368);              // 112,640,000 B
    (void)in_sizes; (void)n_in; (void)out_size; (void)ws_size;

    rgcn_init     <<<(NN * OUT_DIM) / 256, 256, 0, stream>>>(deg, out, bias);
    rgcn_deg      <<<EE / 256,            256, 0, stream>>>(esrc, erel, deg);
    rgcn_pack_feat<<<(NN * KPAD) / 256,   256, 0, stream>>>(features, feat16);
    rgcn_pack_w   <<<901120 / 256,        256, 0, stream>>>(comps, bases, wpack);
    rgcn_gemm     <<<55000 / 8,           256, 0, stream>>>(feat16, wpack, T);
    rgcn_scatter  <<<EE / 8,              256, 0, stream>>>(esrc, erel, edst, deg, T, out);
}
